// dkt_78580721648119
// MI455X (gfx1250) — matrix operand order repaired and checked by reading, not compiled, not run on hardware
//
#include <hip/hip_runtime.h>
#include <hip/hip_bf16.h>

// ---------------- problem constants ----------------
#define S_LEN 100
#define BATCH 1024
#define DHID  64
#define NPROB 6000
#define NTILE_P (NPROB / 16)   // 375

// ---------------- vector types ----------------
typedef __attribute__((ext_vector_type(16))) _Float16 v16h;
typedef __attribute__((ext_vector_type(8)))  _Float16 v8h;
typedef __attribute__((ext_vector_type(8)))  float    v8f;

// ---------------- workspace layout (bytes) ----------------
// packed 7x 64x64 weights (f16 B-layout):      [0,      57344)
// packed out_W (f16 B-layout, 2 kblk x 375):   [57344,  825344)
// out_W transposed [P][64] f32:                [825344, 2361344)
// x_last [1024][64] f32:                       [2361344,2623488)
#define WS_PW7  0
#define WS_POW  57344
#define WS_OWT  825344
#define WS_XL   2361344

__device__ __forceinline__ v8f wmma_f16(v16h a, v16h b, v8f c) {
  return __builtin_amdgcn_wmma_f32_16x16x32_f16(false, a, false, b, (short)0, c,
                                                false, false);
}

__device__ __forceinline__ float sigf(float x) {
  return 1.f / (1.f + __expf(-x));
}
__device__ __forceinline__ float tanhfast(float x) {
  float e = __expf(-2.f * fabsf(x));
  float r = (1.f - e) / (1.f + e);
  return copysignf(r, x);
}

// Build the two 16x32 f16 A operands (K=0..31 and K=32..63) for this lane
// from a row-major [16][64] f16 tile in LDS.
// Lane L<16 holds row L, halves {K0..7, K16..23}; lane L+16 holds {K8..15, K24..31}.
__device__ __forceinline__ void loadA(const _Float16* t, int lr, int hi,
                                      v16h& a0, v16h& a1) {
  v8h c0 = *(const v8h*)(t + lr * 64 +      hi * 8);
  v8h c1 = *(const v8h*)(t + lr * 64 + 16 + hi * 8);
  v8h c2 = *(const v8h*)(t + lr * 64 + 32 + hi * 8);
  v8h c3 = *(const v8h*)(t + lr * 64 + 48 + hi * 8);
  a0 = __builtin_shufflevector(c0, c1, 0,1,2,3,4,5,6,7,8,9,10,11,12,13,14,15);
  a1 = __builtin_shufflevector(c2, c3, 0,1,2,3,4,5,6,7,8,9,10,11,12,13,14,15);
}

// [16,64] x [64,64] matmul: C[nt] = A @ W(w) + bias, 4 n-tiles x 2 WMMAs.
// Packed weight layout: [w][kblk(2)][nblk(4)][lane(32)][16 halves].
__device__ __forceinline__ void mm64(const _Float16* ldsW, int w,
                                     const float* bias, int lane, int lr,
                                     v16h a0, v16h a1, v8f c[4]) {
  const _Float16* base = ldsW + w * 4096 + lane * 16;
#pragma unroll
  for (int nt = 0; nt < 4; ++nt) {
    float bv = bias[nt * 16 + lr];
    v8f acc = {bv, bv, bv, bv, bv, bv, bv, bv};
    v16h b0 = *(const v16h*)(base + (0 * 4 + nt) * 512);
    v16h b1 = *(const v16h*)(base + (1 * 4 + nt) * 512);
    acc = wmma_f16(a0, b0, acc);
    acc = wmma_f16(a1, b1, acc);
    c[nt] = acc;
  }
}

// ---------------- kernel 1: repack weights ----------------
__global__ void dkt_pack(const float* __restrict__ mlpW,
                         const float* __restrict__ Wi,
                         const float* __restrict__ Wh,
                         const float* __restrict__ outW,
                         _Float16* __restrict__ pw7,
                         _Float16* __restrict__ powv,
                         float* __restrict__ owt) {
  const int N7 = 7 * 4096;
  const int NO = 2 * NTILE_P * 512;   // 384000
  const int NT = NPROB * 64;          // 384000
  const int total = N7 + NO + NT;
  for (int t = blockIdx.x * blockDim.x + threadIdx.x; t < total;
       t += gridDim.x * blockDim.x) {
    if (t < N7) {
      int w = t >> 12, rem = t & 4095;
      int kblk = rem >> 11, rem2 = rem & 2047;
      int nblk = rem2 >> 9, rem3 = rem2 & 511;
      int lane = rem3 >> 4, e = rem3 & 15;
      int n = nblk * 16 + (lane & 15);
      int k = kblk * 32 + (e < 8 ? (lane >> 4) * 8 + e : 16 + (lane >> 4) * 8 + (e - 8));
      const float* src = (w == 0) ? mlpW
                       : (w <= 3) ? (Wi + (w - 1) * 4096)
                                  : (Wh + (w - 4) * 4096);
      pw7[t] = (_Float16)src[k * 64 + n];
    } else if (t < N7 + NO) {
      int u = t - N7;
      int kblk = u / (NTILE_P * 512), rem = u % (NTILE_P * 512);
      int nblk = rem / 512, rem3 = rem & 511;
      int lane = rem3 >> 4, e = rem3 & 15;
      int n = nblk * 16 + (lane & 15);
      int k = kblk * 32 + (e < 8 ? (lane >> 4) * 8 + e : 16 + (lane >> 4) * 8 + (e - 8));
      powv[u] = (_Float16)outW[k * NPROB + n];
    } else {
      int u = t - N7 - NO;
      int p = u >> 6, k = u & 63;
      owt[u] = outW[k * NPROB + p];
    }
  }
}

// ---------------- kernel 2: the recurrence ----------------
// One wave32 per 16-row tile; 64 blocks total.
// Dynamic LDS (65280 B):
//   [0,57344)      packed weights (f16)
//   [57344,59136)  biases f32 [7][64]  (mlp_b, bi0..2, bh0..2)
//   [59136,61184)  h tile f16 [16][64]
//   [61184,65280)  union: xm tile f32 [16][64]  /  resp tile f16 [16][64]
__global__ __launch_bounds__(32) void dkt_recur(
    const int* __restrict__ prob_ids, const int* __restrict__ seq_num,
    const float* __restrict__ out_b, const float* __restrict__ prob_emb,
    const float* __restrict__ mlp_b, const float* __restrict__ gru_bi,
    const float* __restrict__ gru_bh, const _Float16* __restrict__ pw7,
    const float* __restrict__ owt, float* __restrict__ xlast,
    float* __restrict__ pred) {
  extern __shared__ char smem[];
  _Float16* ldsW    = (_Float16*)(smem);
  float*    ldsBias = (float*)(smem + 57344);
  _Float16* hTile   = (_Float16*)(smem + 59136);
  float*    xmT     = (float*)(smem + 61184);
  _Float16* respT   = (_Float16*)(smem + 61184);

  const int lane = threadIdx.x;
  const int lr = lane & 15, hi = lane >> 4;
  const int tile = blockIdx.x;
  const int rowBase = tile * 16;

  // stage packed weights + biases into LDS
  for (int i = lane; i < 3584; i += 32)
    ((float4*)ldsW)[i] = ((const float4*)pw7)[i];
  for (int i = lane; i < 64; i += 32)  ldsBias[i]       = mlp_b[i];
  for (int i = lane; i < 192; i += 32) ldsBias[64 + i]  = gru_bi[i];
  for (int i = lane; i < 192; i += 32) ldsBias[256 + i] = gru_bh[i];
  for (int i = lane; i < 128; i += 32)
    ((float4*)hTile)[i] = make_float4(0.f, 0.f, 0.f, 0.f);
  __syncthreads();

  const int seqn = seq_num[rowBase + lr];

  v16h hA0, hA1;
  loadA(hTile, lr, hi, hA0, hA1);
  v8f zero8 = {0.f, 0.f, 0.f, 0.f, 0.f, 0.f, 0.f, 0.f};
  v8f hC[4] = {zero8, zero8, zero8, zero8};

  for (int s = 0; s < S_LEN; ++s) {
    // x = relu(h @ mlp_W + mlp_b)
    v8f xm[4];
    mm64(ldsW, 0, ldsBias, lane, lr, hA0, hA1, xm);
#pragma unroll
    for (int nt = 0; nt < 4; ++nt)
#pragma unroll
      for (int j = 0; j < 8; ++j) {
        float v = xm[nt][j];
        xmT[(hi * 8 + j) * 64 + nt * 16 + lr] = (v > 0.f) ? v : 0.f;
      }
    __syncthreads();

    // prob[b] = x[b] . out_W[:, pid] + out_b[pid]  (lane pair split over K)
    int pid = prob_ids[s * BATCH + rowBase + lr];
    const float* wcol = owt + (size_t)pid * 64 + hi * 32;
    const float* xrow = xmT + lr * 64 + hi * 32;
    float acc = 0.f;
#pragma unroll
    for (int i = 0; i < 8; ++i) {
      float4 xv = ((const float4*)xrow)[i];
      float4 wv = ((const float4*)wcol)[i];
      acc += xv.x * wv.x + xv.y * wv.y + xv.z * wv.z + xv.w * wv.w;
    }
    float prob = acc + __shfl_xor(acc, 16, 32) + out_b[pid];
    if (hi == 0)
      pred[(rowBase + lr) * S_LEN + s] = (s < seqn) ? prob : 0.f;

    if (s == S_LEN - 1) {  // dump x for total_pre_last
      for (int i = lane; i < 1024; i += 32)
        xlast[tile * 1024 + i] = xmT[i];
      break;
    }

    // resp = prob_emb[pid*2 + (prob>0)]  -> f16 tile (aliases xmT; reads done)
    int obs = (prob > 0.f) ? 1 : 0;
    const float* er = prob_emb + ((size_t)pid * 2 + obs) * 64 + hi * 32;
    __syncthreads();
#pragma unroll
    for (int i = 0; i < 32; i += 8) {
      float4 a = ((const float4*)(er + i))[0];
      float4 b = ((const float4*)(er + i))[1];
      v8h hv = {(_Float16)a.x, (_Float16)a.y, (_Float16)a.z, (_Float16)a.w,
                (_Float16)b.x, (_Float16)b.y, (_Float16)b.z, (_Float16)b.w};
      *(v8h*)(respT + lr * 64 + hi * 32 + i) = hv;
    }
    __syncthreads();
    v16h rA0, rA1;
    loadA(respT, lr, hi, rA0, rA1);

    // GRU: 6 matmuls + elementwise
    v8f g1[4], g2[4], rg[4], zg[4];
    mm64(ldsW, 1, ldsBias + 64,  lane, lr, rA0, rA1, g1);
    mm64(ldsW, 4, ldsBias + 256, lane, lr, hA0, hA1, g2);
#pragma unroll
    for (int nt = 0; nt < 4; ++nt)
#pragma unroll
      for (int j = 0; j < 8; ++j) rg[nt][j] = sigf(g1[nt][j] + g2[nt][j]);

    mm64(ldsW, 2, ldsBias + 128, lane, lr, rA0, rA1, g1);
    mm64(ldsW, 5, ldsBias + 320, lane, lr, hA0, hA1, g2);
#pragma unroll
    for (int nt = 0; nt < 4; ++nt)
#pragma unroll
      for (int j = 0; j < 8; ++j) zg[nt][j] = sigf(g1[nt][j] + g2[nt][j]);

    mm64(ldsW, 3, ldsBias + 192, lane, lr, rA0, rA1, g1);
    mm64(ldsW, 6, ldsBias + 384, lane, lr, hA0, hA1, g2);
#pragma unroll
    for (int nt = 0; nt < 4; ++nt)
#pragma unroll
      for (int j = 0; j < 8; ++j) {
        float nv = tanhfast(g1[nt][j] + rg[nt][j] * g2[nt][j]);
        float z = zg[nt][j];
        hC[nt][j] = (1.f - z) * nv + z * hC[nt][j];
      }

    // h -> f16 LDS tile -> A-layout registers for next step
    __syncthreads();
#pragma unroll
    for (int nt = 0; nt < 4; ++nt)
#pragma unroll
      for (int j = 0; j < 8; ++j)
        hTile[(hi * 8 + j) * 64 + nt * 16 + lr] = (_Float16)hC[nt][j];
    __syncthreads();
    loadA(hTile, lr, hi, hA0, hA1);
  }
}

// ---------------- kernel 3: total_pre_last = x_last @ out_W + out_b ----------
__device__ __forceinline__ v8h cvt8(const float* p) {
  float4 a = ((const float4*)p)[0];
  float4 b = ((const float4*)p)[1];
  v8h r = {(_Float16)a.x, (_Float16)a.y, (_Float16)a.z, (_Float16)a.w,
           (_Float16)b.x, (_Float16)b.y, (_Float16)b.z, (_Float16)b.w};
  return r;
}

__global__ __launch_bounds__(32) void dkt_final(
    const _Float16* __restrict__ powv, const float* __restrict__ out_b,
    const float* __restrict__ xlast, float* __restrict__ outp) {
  __shared__ float st[256];
  const int mt = blockIdx.x / NTILE_P;
  const int nt = blockIdx.x % NTILE_P;
  const int lane = threadIdx.x;
  const int lr = lane & 15, hi = lane >> 4;

  const float* xr = xlast + mt * 1024 + lr * 64;
  v8h c0 = cvt8(xr +      hi * 8);
  v8h c1 = cvt8(xr + 16 + hi * 8);
  v8h c2 = cvt8(xr + 32 + hi * 8);
  v8h c3 = cvt8(xr + 48 + hi * 8);
  v16h a0 = __builtin_shufflevector(c0, c1, 0,1,2,3,4,5,6,7,8,9,10,11,12,13,14,15);
  v16h a1 = __builtin_shufflevector(c2, c3, 0,1,2,3,4,5,6,7,8,9,10,11,12,13,14,15);

  const _Float16* bp = powv + (size_t)nt * 512 + lane * 16;
  v16h b0 = *(const v16h*)(bp);
  v16h b1 = *(const v16h*)(bp + NTILE_P * 512);

  float bv = out_b[nt * 16 + lr];
  v8f c = {bv, bv, bv, bv, bv, bv, bv, bv};
  c = wmma_f16(a0, b0, c);
  c = wmma_f16(a1, b1, c);

#pragma unroll
  for (int j = 0; j < 8; ++j) st[(hi * 8 + j) * 16 + lr] = c[j];
  __syncthreads();

  int r = lane >> 1, cb = (lane & 1) * 8;
  float4 v0 = ((const float4*)(st + lane * 8))[0];
  float4 v1 = ((const float4*)(st + lane * 8))[1];
  float* dst = outp + (size_t)(mt * 16 + r) * NPROB + nt * 16 + cb;
  ((float4*)dst)[0] = v0;
  ((float4*)dst)[1] = v1;
}

// ---------------- launch ----------------
extern "C" void kernel_launch(void* const* d_in, const int* in_sizes, int n_in,
                              void* d_out, int out_size, void* d_ws,
                              size_t ws_size, hipStream_t stream) {
  const int*   prob_ids = (const int*)d_in[0];
  const int*   seq_num  = (const int*)d_in[1];
  const float* mlp_W    = (const float*)d_in[2];
  const float* mlp_b    = (const float*)d_in[3];
  const float* out_W    = (const float*)d_in[4];
  const float* out_b    = (const float*)d_in[5];
  const float* gru_Wi   = (const float*)d_in[6];
  const float* gru_bi   = (const float*)d_in[7];
  const float* gru_Wh   = (const float*)d_in[8];
  const float* gru_bh   = (const float*)d_in[9];
  const float* prob_emb = (const float*)d_in[10];

  float* out = (float*)d_out;
  char*  ws  = (char*)d_ws;
  _Float16* pw7  = (_Float16*)(ws + WS_PW7);
  _Float16* powv = (_Float16*)(ws + WS_POW);
  float*    owt  = (float*)(ws + WS_OWT);
  float*    xl   = (float*)(ws + WS_XL);

  dkt_pack<<<1024, 256, 0, stream>>>(mlp_W, gru_Wi, gru_Wh, out_W, pw7, powv, owt);
  dkt_recur<<<BATCH / 16, 32, 65280, stream>>>(prob_ids, seq_num, out_b,
                                               prob_emb, mlp_b, gru_bi, gru_bh,
                                               pw7, owt, xl, out);
  dkt_final<<<(BATCH / 16) * NTILE_P, 32, 0, stream>>>(powv, out_b, xl,
                                                       out + BATCH * S_LEN);
}
